// Model_TimeGatedLSTM_695784702526
// MI455X (gfx1250) — matrix operand order repaired and checked by reading, not compiled, not run on hardware
//
#include <hip/hip_runtime.h>

// ---------------------------------------------------------------------------
// Time-gated 2-layer LSTM for MI455X (gfx1250, wave32, WMMA).
// Recurrent GEMMs use v_wmma_f32_16x16x32_bf16 (bf16 A/B, f32 accum).
// Phase-fused pipeline: one launch per phase p runs layer0(t=p) and
// layer1(t=p-1) concurrently (independent), halving the serial launch chain.
// ---------------------------------------------------------------------------

typedef unsigned short u16;
typedef __bf16 bf16x16 __attribute__((ext_vector_type(16)));
typedef float  f32x8   __attribute__((ext_vector_type(8)));

#define BATCH 256
#define TSEQ  512
#define HID   256
#define BH    (BATCH * HID)      // 65536
#define BT    (BATCH * TSEQ)     // 131072

__device__ __forceinline__ float bf2f(u16 u) {
    unsigned int x = ((unsigned int)u) << 16;
    float f; __builtin_memcpy(&f, &x, 4); return f;
}
__device__ __forceinline__ u16 f2bf(float f) {
    unsigned int x; __builtin_memcpy(&x, &f, 4);
    unsigned int r = x + 0x7fffu + ((x >> 16) & 1u);   // round-to-nearest-even
    return (u16)(r >> 16);
}
__device__ __forceinline__ float sigm(float x) {
    return 1.0f / (1.0f + __expf(-x));
}

// --- f32 [rows, stride] -> bf16 [rows, cols] ---------------------------------
__global__ void conv_bf16_kernel(const float* __restrict__ src, int stride,
                                 u16* __restrict__ dst, int total, int cols) {
    int i = blockIdx.x * blockDim.x + threadIdx.x;
    if (i >= total) return;
    int r = i / cols, c = i - r * cols;
    dst[i] = f2bf(src[(size_t)r * stride + c]);
}

// --- extract x-column (col H) of W0 [4H, H+1] --------------------------------
__global__ void extract_wx_kernel(const float* __restrict__ W0,
                                  float* __restrict__ Wx) {
    int j = blockIdx.x * blockDim.x + threadIdx.x;
    if (j < 4 * HID) Wx[j] = W0[(size_t)j * (HID + 1) + HID];
}

// --- copy initial state ------------------------------------------------------
__global__ void init_state_kernel(const float* __restrict__ h0,
                                  const float* __restrict__ c0,
                                  float* __restrict__ hF0, float* __restrict__ hF1,
                                  u16* __restrict__ hB0, u16* __restrict__ hB1,
                                  float* __restrict__ cb0, float* __restrict__ cb1) {
    int i = blockIdx.x * blockDim.x + threadIdx.x;
    if (i >= BH) return;
    float a = h0[i], b = h0[BH + i];
    hF0[i] = a; hF1[i] = b;
    hB0[i] = f2bf(a); hB1[i] = f2bf(b);
    cb0[i] = c0[i]; cb1[i] = c0[BH + i];
}

// --- one recurrent timestep of one layer (device body) -----------------------
// Block = 128 threads = 4 waves. Wave gi computes the 16x16 tile of gate gi
// (gate columns gi*H + nblock*16 .. +15, batch rows mtile*16 .. +15) with
// WMMA bf16 over K = KSTEPS*32. LDS exchange, then fused cell update.
union Frag { uint4 u[2]; bf16x16 v; };

template <int KSTEPS, bool L0>
__device__ __forceinline__
void tg_step_body(float (*lds)[16][16],                // [4][16][16] shared
                  const u16*  __restrict__ Wb,         // [4H, KSTEPS*32] bf16
                  const u16*  __restrict__ hin,        // [B, H] bf16 (prev h)
                  const u16*  __restrict__ xin,        // [B, H] bf16 (layer-1 input)
                  const float* __restrict__ Xg,        // X as [B, T]
                  const float* __restrict__ bias,      // [4H]
                  const float* __restrict__ Wx,        // [4H] (layer 0) or null
                  const float* __restrict__ Wt,        // [3H]
                  const float* __restrict__ bt,        // [3H]
                  float* __restrict__ cbuf,            // [B, H]
                  float* __restrict__ hout_f,          // [B, H]
                  u16*   __restrict__ hout_bf,         // [B, H] (next parity)
                  u16*   __restrict__ yout,            // y[t] -> [B, H]
                  int t) {
    constexpr int Kw = KSTEPS * 32;

    const int tid    = threadIdx.x;
    const int lane   = tid & 31;
    const int gi     = tid >> 5;          // gate index 0..3
    const int mtile  = blockIdx.x;        // 0..15 (batch tiles)
    const int nblock = blockIdx.y;        // 0..15 (H column tiles)

    const int m    = (mtile << 4) + (lane & 15);              // batch row
    const int g    = lane >> 4;                               // lane half
    const int nrow = gi * HID + (nblock << 4) + (lane & 15);  // weight row / out col

    f32x8 acc = {0.f, 0.f, 0.f, 0.f, 0.f, 0.f, 0.f, 0.f};

#pragma unroll
    for (int kk = 0; kk < KSTEPS; ++kk) {
        const u16* asrc = hin;
        int ka = kk * 32;
        if constexpr (KSTEPS > 8) {
            if (kk >= 8) { asrc = xin; ka = kk * 32 - HID; }
        }
        Frag a, b;
        // A 16x32 bf16: lanes 0-15 hold K = ka+0..7 & ka+16..23 (g=0);
        //               lanes 16-31 hold K = ka+8..15 & ka+24..31 (g=1).
        a.u[0] = *reinterpret_cast<const uint4*>(asrc + (size_t)m * HID + ka + g * 8);
        a.u[1] = *reinterpret_cast<const uint4*>(asrc + (size_t)m * HID + ka + 16 + g * 8);
        // B 32x16 bf16: lane n holds K = g*8..g*8+7 & 16+g*8..16+g*8+7 of W row n.
        const u16* bp = Wb + (size_t)nrow * Kw + kk * 32 + g * 8;
        b.u[0] = *reinterpret_cast<const uint4*>(bp);
        b.u[1] = *reinterpret_cast<const uint4*>(bp + 16);
        acc = __builtin_amdgcn_wmma_f32_16x16x32_bf16(
            false, a.v, false, b.v, (short)0, acc, false, false);
    }

    // D layout: lane l -> col n = l&15; VGPR v -> row (l>>4)*8 + v.
    {
        const int nn = lane & 15;
        const int mb = (lane >> 4) * 8;
#pragma unroll
        for (int v = 0; v < 8; ++v) lds[gi][mb + v][nn] = acc[v];
    }
    __syncthreads();

    // Fused LSTM cell + time-gate update: 256 elements / 128 threads.
    for (int e = tid; e < 256; e += 128) {
        const int mm = e >> 4, nn = e & 15;
        const int b  = (mtile << 4) + mm;          // batch
        const int jg = (nblock << 4) + nn;         // H index
        const float x_tb = Xg[(size_t)b * TSEQ + t];

        float ig = lds[0][mm][nn] + bias[jg];
        float fg = lds[1][mm][nn] + bias[HID + jg];
        float cg = lds[2][mm][nn] + bias[2 * HID + jg];
        float og = lds[3][mm][nn] + bias[3 * HID + jg];
        if constexpr (L0) {
            ig += Wx[jg] * x_tb;
            fg += Wx[HID + jg] * x_tb;
            cg += Wx[2 * HID + jg] * x_tb;
            og += Wx[3 * HID + jg] * x_tb;
        }
        const float itg = sigm(Wt[jg] * x_tb + bt[jg]);
        const float ftg = sigm(Wt[HID + jg] * x_tb + bt[HID + jg]);
        const float otg = sigm(Wt[2 * HID + jg] * x_tb + bt[2 * HID + jg]);

        const size_t off = (size_t)b * HID + jg;
        const float c_old = cbuf[off];
        const float nc = c_old * sigm(fg) * ftg + tanhf(cg) * sigm(ig) * itg;
        const float nh = tanhf(nc) * sigm(og) * otg;
        cbuf[off]   = nc;
        hout_f[off] = nh;
        const u16 nb = f2bf(nh);
        hout_bf[off] = nb;
        yout[off]    = nb;
    }
}

// --- fused phase kernel: z=0 -> layer0(t=p), z=1 -> layer1(t=p-1) ------------
struct StepArgs {
    const u16*   W0h;
    const u16*   W1b;
    const float* Xg;
    const float* b0;
    const float* Wx0;
    const float* Wt0;
    const float* bt0;
    const float* b1;
    const float* Wt1;
    const float* bt1;
    float* cb0;
    float* cb1;
    float* hF0;
    float* hF1;
    u16*   hB0a; u16* hB0b;
    u16*   hB1a; u16* hB1b;
    u16*   y0;
    u16*   y1;
};

__global__ __launch_bounds__(128)
void tg_phase_kernel(StepArgs a, int p) {
    __shared__ float lds[4][16][16];
    if (blockIdx.z == 0) {
        if (p >= TSEQ) return;
        const int t = p;
        u16* hin  = (t & 1) ? a.hB0b : a.hB0a;
        u16* hout = (t & 1) ? a.hB0a : a.hB0b;
        tg_step_body<8, true>(lds, a.W0h, hin, nullptr, a.Xg,
                              a.b0, a.Wx0, a.Wt0, a.bt0,
                              a.cb0, a.hF0, hout, a.y0 + (size_t)t * BH, t);
    } else {
        const int t = p - 1;
        if (t < 0) return;
        u16* hin  = (t & 1) ? a.hB1b : a.hB1a;
        u16* hout = (t & 1) ? a.hB1a : a.hB1b;
        tg_step_body<16, false>(lds, a.W1b, hin, a.y0 + (size_t)t * BH, a.Xg,
                                a.b1, nullptr, a.Wt1, a.bt1,
                                a.cb1, a.hF1, hout, a.y1 + (size_t)t * BH, t);
    }
}

// --- final FC: out[b*T+t] = y1[t,b,:] . fc_w + fc_b --------------------------
__global__ void fc_kernel(const u16* __restrict__ y1,
                          const float* __restrict__ w,
                          const float* __restrict__ fb,
                          float* __restrict__ out) {
    int idx = blockIdx.x * blockDim.x + threadIdx.x;   // 0..BT-1 = b*T + t
    if (idx >= BT) return;
    int b = idx >> 9;          // / TSEQ
    int t = idx & (TSEQ - 1);
    const u16* yp = y1 + (size_t)t * BH + (size_t)b * HID;
    float s = fb[0];
#pragma unroll 8
    for (int j = 0; j < HID; ++j) s += bf2f(yp[j]) * w[j];
    out[idx] = s;
}

// --- copy final h/c states into d_out ----------------------------------------
__global__ void finalize_kernel(const float* __restrict__ hF0,
                                const float* __restrict__ hF1,
                                const float* __restrict__ cb0,
                                const float* __restrict__ cb1,
                                float* __restrict__ out) {
    int i = blockIdx.x * blockDim.x + threadIdx.x;
    if (i >= BH) return;
    out[BT + i]            = hF0[i];
    out[BT + BH + i]       = hF1[i];
    out[BT + 2 * BH + i]   = cb0[i];
    out[BT + 3 * BH + i]   = cb1[i];
}

// ---------------------------------------------------------------------------
extern "C" void kernel_launch(void* const* d_in, const int* in_sizes, int n_in,
                              void* d_out, int out_size, void* d_ws, size_t ws_size,
                              hipStream_t stream) {
    const float* X   = (const float*)d_in[0];   // [B,T,1]
    const float* h0  = (const float*)d_in[1];   // [2,B,H]
    const float* c0  = (const float*)d_in[2];   // [2,B,H]
    const float* W0  = (const float*)d_in[3];   // [4H, H+1]
    const float* b0  = (const float*)d_in[4];   // [4H]
    const float* Wt0 = (const float*)d_in[5];   // [3H]
    const float* bt0 = (const float*)d_in[6];   // [3H]
    const float* W1  = (const float*)d_in[7];   // [4H, 2H]
    const float* b1  = (const float*)d_in[8];   // [4H]
    const float* Wt1 = (const float*)d_in[9];   // [3H]
    const float* bt1 = (const float*)d_in[10];  // [3H]
    const float* fcw = (const float*)d_in[11];  // [H]
    const float* fcb = (const float*)d_in[12];  // [1]
    float* out = (float*)d_out;

    // ---- workspace carve-up (all offsets 256B aligned) ----
    char*  ws = (char*)d_ws;
    size_t o  = 0;
    auto carve = [&](size_t bytes) -> void* {
        void* p = ws + o;
        o += (bytes + 255) & ~(size_t)255;
        return p;
    };
    u16*   W0h   = (u16*)carve((size_t)4 * HID * HID * 2);        // [4H, H] bf16
    u16*   W1b   = (u16*)carve((size_t)4 * HID * 2 * HID * 2);    // [4H, 2H] bf16
    float* Wx0   = (float*)carve((size_t)4 * HID * 4);
    float* hF0   = (float*)carve((size_t)BH * 4);
    float* hF1   = (float*)carve((size_t)BH * 4);
    float* cb0   = (float*)carve((size_t)BH * 4);
    float* cb1   = (float*)carve((size_t)BH * 4);
    u16*   hB0a  = (u16*)carve((size_t)BH * 2);
    u16*   hB0b  = (u16*)carve((size_t)BH * 2);
    u16*   hB1a  = (u16*)carve((size_t)BH * 2);
    u16*   hB1b  = (u16*)carve((size_t)BH * 2);
    u16*   y0    = (u16*)carve((size_t)TSEQ * BH * 2);            // layer-0 outputs
    u16*   y1    = (u16*)carve((size_t)TSEQ * BH * 2);            // layer-1 outputs

    // ---- weight prep + state init ----
    {
        int tot0 = 4 * HID * HID;          // 262144
        conv_bf16_kernel<<<(tot0 + 255) / 256, 256, 0, stream>>>(W0, HID + 1, W0h, tot0, HID);
        int tot1 = 4 * HID * 2 * HID;      // 524288
        conv_bf16_kernel<<<(tot1 + 255) / 256, 256, 0, stream>>>(W1, 2 * HID, W1b, tot1, 2 * HID);
        extract_wx_kernel<<<(4 * HID + 255) / 256, 256, 0, stream>>>(W0, Wx0);
        init_state_kernel<<<(BH + 255) / 256, 256, 0, stream>>>(h0, c0, hF0, hF1,
                                                                hB0a, hB1a, cb0, cb1);
    }

    StepArgs sa;
    sa.W0h = W0h; sa.W1b = W1b; sa.Xg = X;
    sa.b0 = b0; sa.Wx0 = Wx0; sa.Wt0 = Wt0; sa.bt0 = bt0;
    sa.b1 = b1; sa.Wt1 = Wt1; sa.bt1 = bt1;
    sa.cb0 = cb0; sa.cb1 = cb1; sa.hF0 = hF0; sa.hF1 = hF1;
    sa.hB0a = hB0a; sa.hB0b = hB0b; sa.hB1a = hB1a; sa.hB1b = hB1b;
    sa.y0 = y0; sa.y1 = y1;

    // ---- fused phase pipeline: p runs layer0(t=p) and layer1(t=p-1) ----
    const dim3 pgrid(16, 16, 2);
    for (int p = 0; p <= TSEQ; ++p) {
        tg_phase_kernel<<<pgrid, 128, 0, stream>>>(sa, p);
    }

    // ---- FC head + final states ----
    fc_kernel<<<(BT + 255) / 256, 256, 0, stream>>>(y1, fcw, fcb, out);
    finalize_kernel<<<(BH + 255) / 256, 256, 0, stream>>>(hF0, hF1, cb0, cb1, out);
}
